// MixiTBlock_21930103014123
// MI455X (gfx1250) — matrix operand order repaired and checked by reading, not compiled, not run on hardware
//
#include <hip/hip_runtime.h>

// ---------------------------------------------------------------------------
// MixiT/CoaT block on gfx1250 (MI455X). BF16 WMMA GEMMs (f32 accum), fused
// epilogues, fp32 VALU for LN/softmax/dwconv/ktv. GEMM staging uses CDNA5
// async global->LDS copies when the builtin is available.
// ---------------------------------------------------------------------------

typedef unsigned short u16;
typedef __bf16 bf16_t;
typedef bf16_t v16bf __attribute__((ext_vector_type(16)));
typedef float  v8f   __attribute__((ext_vector_type(8)));
typedef int    v4i   __attribute__((vector_size(16)));

#define BB   32
#define HH   32
#define WW   32
#define NN   1024
#define CC   512
#define NHEAD 8
#define DD   64
#define CHID 2048
#define MM   (BB * NN)   // 32768 tokens

#if defined(__has_builtin)
#if __has_builtin(__builtin_amdgcn_global_load_async_to_lds_b128)
#define USE_ASYNC_LDS 1
#endif
#endif
#ifndef USE_ASYNC_LDS
#define USE_ASYNC_LDS 0
#endif

// ---------------- bf16 helpers --------------------------------------------
static __device__ __forceinline__ u16 f2bf(float f) {
    unsigned int u = __float_as_uint(f);
    unsigned int r = u + 0x7FFFu + ((u >> 16) & 1u);
    return (u16)(r >> 16);
}
static __device__ __forceinline__ float bf2f(u16 h) {
    return __uint_as_float(((unsigned int)h) << 16);
}
static __device__ __forceinline__ float gelu_exact(float v) {
    return 0.5f * v * (1.0f + erff(v * 0.70710678118654752f));
}

// 16B global -> LDS copy: async on CDNA5 (ASYNCcnt), else via VGPRs ---------
static __device__ __forceinline__ void cp16(void* l, const void* g) {
#if USE_ASYNC_LDS
    __builtin_amdgcn_global_load_async_to_lds_b128(
        (__attribute__((address_space(1))) v4i*)g,
        (__attribute__((address_space(3))) v4i*)l, 0, 0);
#else
    *(uint4*)l = *(const uint4*)g;
#endif
}
static __device__ __forceinline__ void cp16_wait() {
#if USE_ASYNC_LDS
    asm volatile("s_wait_asynccnt 0" ::: "memory");
#endif
}

// vec4 load/store (float or bf16 buffers) ----------------------------------
static __device__ __forceinline__ void load4f(const float* p, float* f) {
    const float4 t = *(const float4*)p;
    f[0] = t.x; f[1] = t.y; f[2] = t.z; f[3] = t.w;
}
static __device__ __forceinline__ void load4f(const u16* p, float* f) {
    const uint2 t = *(const uint2*)p;
    f[0] = bf2f((u16)(t.x & 0xffff)); f[1] = bf2f((u16)(t.x >> 16));
    f[2] = bf2f((u16)(t.y & 0xffff)); f[3] = bf2f((u16)(t.y >> 16));
}
static __device__ __forceinline__ void store4f(float* p, const float* f) {
    *(float4*)p = make_float4(f[0], f[1], f[2], f[3]);
}
static __device__ __forceinline__ void store4f(u16* p, const float* f) {
    uint2 t;
    t.x = (unsigned)f2bf(f[0]) | ((unsigned)f2bf(f[1]) << 16);
    t.y = (unsigned)f2bf(f[2]) | ((unsigned)f2bf(f[3]) << 16);
    *(uint2*)p = t;
}

// ---------------- weight fp32 -> bf16 -------------------------------------
__global__ __launch_bounds__(256) void cvt_bf16_k(const float* __restrict__ in,
                                                  u16* __restrict__ out, int n) {
    int i = (blockIdx.x * 256 + threadIdx.x) * 4;
    if (i >= n) return;
    float f[4] = { in[i], in[i + 1], in[i + 2], in[i + 3] };
    store4f(out + i, f);
}

// ---------------- depthwise 3x3 conv (SAME), optional residual ------------
template <typename InT, typename OutT>
__global__ __launch_bounds__(256) void dwconv3x3_k(
    const InT* __restrict__ in, size_t inStride,
    const InT* __restrict__ res, size_t resStride,
    OutT* __restrict__ out, size_t outStride,
    const float* __restrict__ w, const float* __restrict__ bias, int C) {
    const int cv = C >> 2;
    size_t idx = (size_t)blockIdx.x * 256 + threadIdx.x;
    size_t total = (size_t)BB * NN * cv;
    if (idx >= total) return;
    int c4 = (int)(idx % cv);
    size_t rest = idx / cv;
    int n = (int)(rest % NN);
    int b = (int)(rest / NN);
    int y = n >> 5, x = n & 31;
    int c0 = c4 * 4;

    float acc[4] = {0.f, 0.f, 0.f, 0.f};
    #pragma unroll
    for (int ky = 0; ky < 3; ++ky) {
        int yy = y + ky - 1;
        if (yy < 0 || yy >= HH) continue;
        #pragma unroll
        for (int kx = 0; kx < 3; ++kx) {
            int xx = x + kx - 1;
            if (xx < 0 || xx >= WW) continue;
            float f[4];
            load4f(in + ((size_t)b * NN + yy * WW + xx) * inStride + c0, f);
            int k = ky * 3 + kx;
            #pragma unroll
            for (int j = 0; j < 4; ++j)
                acc[j] = fmaf(f[j], w[(size_t)(c0 + j) * 9 + k], acc[j]);
        }
    }
    #pragma unroll
    for (int j = 0; j < 4; ++j) acc[j] += bias[c0 + j];
    if (res) {
        float f[4];
        load4f(res + ((size_t)b * NN + n) * resStride + c0, f);
        #pragma unroll
        for (int j = 0; j < 4; ++j) acc[j] += f[j];
    }
    store4f(out + ((size_t)b * NN + n) * outStride + c0, acc);
}

// ---------------- LayerNorm over C=512 -> bf16 ----------------------------
__global__ __launch_bounds__(128) void ln_k(const float* __restrict__ in,
                                            const float* __restrict__ g,
                                            const float* __restrict__ be,
                                            u16* __restrict__ out) {
    const size_t tok = blockIdx.x;
    const int tid = threadIdx.x;
    const int lane = tid & 31, wid = tid >> 5;
    const float* p = in + tok * CC + tid * 4;
    float4 v = *(const float4*)p;
    float s1 = v.x + v.y + v.z + v.w;
    float s2 = v.x * v.x + v.y * v.y + v.z * v.z + v.w * v.w;
    #pragma unroll
    for (int off = 16; off > 0; off >>= 1) {
        s1 += __shfl_xor(s1, off, 32);
        s2 += __shfl_xor(s2, off, 32);
    }
    __shared__ float a1[4], a2[4];
    if (lane == 0) { a1[wid] = s1; a2[wid] = s2; }
    __syncthreads();
    s1 = a1[0] + a1[1] + a1[2] + a1[3];
    s2 = a2[0] + a2[1] + a2[2] + a2[3];
    float mu = s1 * (1.0f / CC);
    float var = s2 * (1.0f / CC) - mu * mu;
    float rstd = rsqrtf(var + 1e-6f);
    int c = tid * 4;
    float f[4] = { (v.x - mu) * rstd * g[c]     + be[c],
                   (v.y - mu) * rstd * g[c + 1] + be[c + 1],
                   (v.z - mu) * rstd * g[c + 2] + be[c + 2],
                   (v.w - mu) * rstd * g[c + 3] + be[c + 3] };
    store4f(out + tok * CC + c, f);
}

// ---------------- BF16 WMMA GEMM: out = act(A * W^T + bias) [+res] --------
// A: M x K bf16 row-major; Wt: O x K bf16 row-major. Block tile 64x128,
// 8 waves (4 row-groups x 2 col-groups), each wave 16x64 via 4 WMMA tiles.
#define LDA 40
#define LDB 40
__global__ __launch_bounds__(256) void gemm_bf16_k(
    const u16* __restrict__ A, const u16* __restrict__ Wt,
    const float* __restrict__ bias, const float* __restrict__ resid,
    float* __restrict__ outF, u16* __restrict__ outB,
    int M, int K, int O, int actGelu) {
    __shared__ u16 lA[64 * LDA];
    __shared__ u16 lB[128 * LDB];
    const int tid = threadIdx.x;
    const int lane = tid & 31, wid = tid >> 5;
    const int wm = wid & 3, wn = wid >> 2;
    const int half = lane >> 4, l16 = lane & 15;
    const int n0 = blockIdx.x * 128;
    const int m0 = blockIdx.y * 64;

    v8f acc[4] = {};

    for (int k0 = 0; k0 < K; k0 += 32) {
        // cooperative stage: A tile 64x32, B tile 128x32 (weights)
        {
            int row = tid >> 2, chunk = tid & 3;
            cp16(&lA[row * LDA + chunk * 8],
                 &A[(size_t)(m0 + row) * K + k0 + chunk * 8]);
            if (k0 + 32 < K)
                __builtin_prefetch(&A[(size_t)(m0 + row) * K + k0 + 32], 0, 1);
        }
        #pragma unroll
        for (int i = 0; i < 2; ++i) {
            int idx = tid + i * 256;
            int row = idx >> 2, chunk = idx & 3;
            cp16(&lB[row * LDB + chunk * 8],
                 &Wt[(size_t)(n0 + row) * K + k0 + chunk * 8]);
        }
        cp16_wait();
        __syncthreads();

        // A fragment: K = 16*half + e (matches B)
        v16bf af;
        {
            const u16* p = &lA[(wm * 16 + l16) * LDA];
            ((uint4*)&af)[0] = *(const uint4*)(p + 16 * half);
            ((uint4*)&af)[1] = *(const uint4*)(p + 16 * half + 8);
        }
        #pragma unroll
        for (int s = 0; s < 4; ++s) {
            v16bf bfg;   // B fragment: K = 16*half + e, column = l16
            const u16* p = &lB[(wn * 64 + s * 16 + l16) * LDB + 16 * half];
            ((uint4*)&bfg)[0] = *(const uint4*)(p);
            ((uint4*)&bfg)[1] = *(const uint4*)(p + 8);
            acc[s] = __builtin_amdgcn_wmma_f32_16x16x32_bf16(
                false, af, false, bfg, (short)0, acc[s], false, false);
        }
        __syncthreads();
    }

    // epilogue: C/D layout row = vg + 8*half, col = l16
    #pragma unroll
    for (int s = 0; s < 4; ++s) {
        int col = n0 + wn * 64 + s * 16 + l16;
        float bv = bias ? bias[col] : 0.f;
        #pragma unroll
        for (int vg = 0; vg < 8; ++vg) {
            int row = m0 + wm * 16 + vg + 8 * half;
            float v = acc[s][vg] + bv;
            if (actGelu) v = gelu_exact(v);
            size_t o = (size_t)row * O + col;
            if (resid) v += resid[o];
            if (outF) outF[o] = v;
            if (outB) outB[o] = f2bf(v);
        }
    }
}

// ---------------- softmax over tokens for k -------------------------------
// block = (hc-group of 32, batch b); lanes -> channels (coalesced)
__global__ __launch_bounds__(256) void softmax_k(const float* __restrict__ qkv,
                                                 float* __restrict__ ks) {
    const int tid = threadIdx.x;
    const int lane = tid & 31, wid = tid >> 5;
    const int hc = blockIdx.x * 32 + lane;
    const int b = blockIdx.y;
    const size_t base = ((size_t)b * NN) * (3 * CC) + CC + hc;
    __shared__ float red[8 * 32];

    float mx = -3.4e38f;
    for (int i = 0; i < 128; ++i)
        mx = fmaxf(mx, qkv[base + (size_t)(wid + 8 * i) * (3 * CC)]);
    red[wid * 32 + lane] = mx;
    __syncthreads();
    float m = red[lane];
    #pragma unroll
    for (int j = 1; j < 8; ++j) m = fmaxf(m, red[j * 32 + lane]);
    __syncthreads();

    float sm = 0.f;
    for (int i = 0; i < 128; ++i) {
        int n = wid + 8 * i;
        float e = __expf(qkv[base + (size_t)n * (3 * CC)] - m);
        ks[((size_t)b * NN + n) * CC + hc] = e;
        sm += e;
    }
    red[wid * 32 + lane] = sm;
    __syncthreads();
    float s = red[lane];
    #pragma unroll
    for (int j = 1; j < 8; ++j) s += red[j * 32 + lane];
    float inv = 1.0f / s;
    for (int i = 0; i < 128; ++i) {
        int n = wid + 8 * i;
        ks[((size_t)b * NN + n) * CC + hc] *= inv;
    }
}

// ---------------- ktv = k_softmax^T v, stored transposed as bf16 ----------
// one block per (b, head); LDS-tiled over 64 tokens at a time
__global__ __launch_bounds__(256) void ktv_k(const float* __restrict__ ks,
                                             const float* __restrict__ qkv,
                                             u16* __restrict__ ktvT) {
    const int bh = blockIdx.x;          // b*8 + h
    const int b = bh >> 3, h = bh & 7;
    const int tid = threadIdx.x;
    const int kd = tid & 63;
    const int vdb = tid >> 6;           // 0..3
    __shared__ float lK[64 * 64];
    __shared__ float lV[64 * 64];

    float acc[16];
    #pragma unroll
    for (int i = 0; i < 16; ++i) acc[i] = 0.f;

    for (int t0 = 0; t0 < NN; t0 += 64) {
        #pragma unroll
        for (int i = 0; i < 16; ++i) {
            int e = tid + 256 * i;
            int row = e >> 6, col = e & 63;
            lK[e] = ks[((size_t)b * NN + t0 + row) * CC + h * DD + col];
            lV[e] = qkv[((size_t)b * NN + t0 + row) * (3 * CC) + 2 * CC + h * DD + col];
        }
        __syncthreads();
        for (int nn = 0; nn < 64; ++nn) {
            float kv = lK[nn * 64 + kd];
            #pragma unroll
            for (int i = 0; i < 16; ++i)
                acc[i] = fmaf(kv, lV[nn * 64 + vdb + 4 * i], acc[i]);
        }
        __syncthreads();
    }
    #pragma unroll
    for (int i = 0; i < 16; ++i) {
        int vd = vdb + 4 * i;
        ktvT[(((size_t)bh * DD) + vd) * DD + kd] = f2bf(acc[i]);
    }
}

// ---------------- factor_att (WMMA) + crpe + residual ---------------------
// grid: (token-group of 128, head, batch); wave = 16 tokens x 64 vd
__global__ __launch_bounds__(256) void attn_k(
    const u16* __restrict__ qbf, const float* __restrict__ qkvF,
    const u16* __restrict__ ktvT, const float* __restrict__ vconv,
    const float* __restrict__ x1, float* __restrict__ x2) {
    const int tid = threadIdx.x;
    const int lane = tid & 31, wid = tid >> 5;
    const int half = lane >> 4, l16 = lane & 15;
    const int tok0 = blockIdx.x * 128 + wid * 16;
    const int h = blockIdx.y, b = blockIdx.z;
    const size_t bh = (size_t)b * NHEAD + h;

    v8f acc[4] = {};
    #pragma unroll
    for (int kk = 0; kk < 2; ++kk) {
        int k0 = kk * 32;
        v16bf af;   // q fragment: token rows, kd columns
        {
            const u16* p = qbf + ((size_t)b * NN + tok0 + l16) * (3 * CC) + h * DD + k0;
            ((uint4*)&af)[0] = *(const uint4*)(p + 16 * half);
            ((uint4*)&af)[1] = *(const uint4*)(p + 16 * half + 8);
        }
        #pragma unroll
        for (int s = 0; s < 4; ++s) {
            int vd = s * 16 + l16;
            v16bf bfg;  // ktvT[vd][kd] -> B[k][n]
            const u16* p = ktvT + (bh * DD + vd) * DD + k0 + 16 * half;
            ((uint4*)&bfg)[0] = *(const uint4*)(p);
            ((uint4*)&bfg)[1] = *(const uint4*)(p + 8);
            acc[s] = __builtin_amdgcn_wmma_f32_16x16x32_bf16(
                false, af, false, bfg, (short)0, acc[s], false, false);
        }
    }

    const float scale = 0.125f;  // d^-0.5, d=64
    #pragma unroll
    for (int s = 0; s < 4; ++s) {
        #pragma unroll
        for (int vg = 0; vg < 8; ++vg) {
            int token = tok0 + vg + 8 * half;
            int c = h * DD + s * 16 + l16;
            size_t idx = ((size_t)b * NN + token) * CC + c;
            size_t qidx = ((size_t)b * NN + token) * (3 * CC) + c;
            x2[idx] = x1[idx] + scale * acc[s][vg] + qkvF[qidx] * vconv[idx];
        }
    }
}

// ---------------------------------------------------------------------------
extern "C" void kernel_launch(void* const* d_in, const int* in_sizes, int n_in,
                              void* d_out, int out_size, void* d_ws, size_t ws_size,
                              hipStream_t stream) {
    const float* x      = (const float*)d_in[0];
    const float* cpe_w  = (const float*)d_in[1];
    const float* cpe_b  = (const float*)d_in[2];
    const float* ln1_w  = (const float*)d_in[3];
    const float* ln1_b  = (const float*)d_in[4];
    const float* qkv_w  = (const float*)d_in[5];
    const float* crpe_w = (const float*)d_in[6];
    const float* crpe_b = (const float*)d_in[7];
    const float* ln2_w  = (const float*)d_in[8];
    const float* ln2_b  = (const float*)d_in[9];
    const float* fc1_w  = (const float*)d_in[10];
    const float* fc1_b  = (const float*)d_in[11];
    const float* dw_w   = (const float*)d_in[12];
    const float* dw_b   = (const float*)d_in[13];
    const float* fc2_w  = (const float*)d_in[14];
    const float* fc2_b  = (const float*)d_in[15];
    float* out = (float*)d_out;

    char* wsb = (char*)d_ws;
    // phase-reused workspace layout (peak ~612 MB)
    float* x1    = (float*)(wsb);                              // 64 MB
    u16*   xn    = (u16*)(wsb + 67108864ull);                  // 32 MB (-> x2n later)
    float* qkv   = (float*)(wsb + 100663296ull);               // 192 MB (-> hdn later)
    u16*   qkvb  = (u16*)(wsb + 301989888ull);                 // 96 MB (-> hdn2 later)
    float* ks    = (float*)(wsb + 402653184ull);               // 64 MB
    u16*   ktvT  = (u16*)(wsb + 469762048ull);                 // 2 MB
    float* vconv = (float*)(wsb + 471859200ull);               // 64 MB
    float* x2    = (float*)(wsb + 538968064ull);               // 64 MB
    u16*   wqb   = (u16*)(wsb + 606076928ull);                 // 1536x512 bf16
    u16*   wf1b  = wqb + 1536 * 512;                           // 2048x512 bf16
    u16*   wf2b  = wf1b + 1048576;                             // 512x2048 bf16
    u16*   hdn   = (u16*)(wsb + 100663296ull);                 // alias qkv (dead)
    u16*   hdn2  = (u16*)(wsb + 301989888ull);                 // alias qkvb+ks (dead)
    u16*   x2n   = xn;                                         // alias xn (dead)

    // 1) weights -> bf16
    cvt_bf16_k<<<(1536 * 512) / 1024, 256, 0, stream>>>(qkv_w, wqb, 1536 * 512);
    cvt_bf16_k<<<(2048 * 512) / 1024, 256, 0, stream>>>(fc1_w, wf1b, 2048 * 512);
    cvt_bf16_k<<<(512 * 2048) / 1024, 256, 0, stream>>>(fc2_w, wf2b, 512 * 2048);

    // 2) CPE: x1 = x + dwconv(x)
    dwconv3x3_k<float, float><<<(BB * NN * (CC / 4)) / 256, 256, 0, stream>>>(
        x, CC, x, CC, x1, CC, cpe_w, cpe_b, CC);

    // 3) LN1 -> bf16
    ln_k<<<MM, 128, 0, stream>>>(x1, ln1_w, ln1_b, xn);

    // 4) qkv GEMM (fp32 + bf16 copy)
    gemm_bf16_k<<<dim3(1536 / 128, MM / 64), 256, 0, stream>>>(
        xn, wqb, nullptr, nullptr, qkv, qkvb, MM, CC, 3 * CC, 0);

    // 5) softmax over tokens for k
    softmax_k<<<dim3(CC / 32, BB), 256, 0, stream>>>(qkv, ks);

    // 6) crpe dwconv on v (no residual)
    dwconv3x3_k<float, float><<<(BB * NN * (CC / 4)) / 256, 256, 0, stream>>>(
        qkv + 2 * CC, 3 * CC, (const float*)nullptr, 0, vconv, CC, crpe_w, crpe_b, CC);

    // 7) ktv (transposed bf16)
    ktv_k<<<BB * NHEAD, 256, 0, stream>>>(ks, qkv, ktvT);

    // 8) x2 = x1 + scale*q@ktv + q*vconv   (WMMA)
    attn_k<<<dim3(NN / 128, NHEAD, BB), 256, 0, stream>>>(
        qkvb, qkv, ktvT, vconv, x1, x2);

    // 9) LN2 -> bf16
    ln_k<<<MM, 128, 0, stream>>>(x2, ln2_w, ln2_b, x2n);

    // 10) fc1 GEMM + bias + exact GELU -> bf16 hidden
    gemm_bf16_k<<<dim3(CHID / 128, MM / 64), 256, 0, stream>>>(
        x2n, wf1b, fc1_b, nullptr, nullptr, hdn, MM, CC, CHID, 1);

    // 11) hidden dwconv + residual (bf16)
    dwconv3x3_k<u16, u16><<<(BB * NN * (CHID / 4)) / 256, 256, 0, stream>>>(
        hdn, CHID, hdn, CHID, hdn2, CHID, dw_w, dw_b, CHID);

    // 12) fc2 GEMM + bias + residual(x2) -> out
    gemm_bf16_k<<<dim3(CC / 128, MM / 64), 256, 0, stream>>>(
        hdn2, wf2b, fc2_b, x2, out, nullptr, MM, CHID, CC, 0);
}
